// GAT_10943576670341
// MI455X (gfx1250) — matrix operand order repaired and checked by reading, not compiled, not run on hardware
//
#include <hip/hip_runtime.h>
#include <hip/hip_bf16.h>

// ---------------- problem constants ----------------
#define HEADS   56
#define OUTD    32
#define NFEAT   128
#define D1      1792          // HEADS*OUTD
#define NNODES  20000
#define NEDGES  100000
#define NGRAPH  50
#define ETOT    (NEDGES + NNODES)   // edges + self loops = 120000
#define EPS_BN  1e-5f

typedef __attribute__((ext_vector_type(16))) __bf16 v16bf;
typedef __attribute__((ext_vector_type(8)))  float  v8f;
typedef unsigned int u32x4 __attribute__((ext_vector_type(4)));
typedef int          i32x4 __attribute__((ext_vector_type(4)));
typedef int          i32x8 __attribute__((ext_vector_type(8)));

#if defined(__has_builtin)
#if __has_builtin(__builtin_amdgcn_tensor_load_to_lds)
#define USE_TDM 1
#endif
#endif

// ---------------- small helpers ----------------
__device__ __forceinline__ float lrelu02(float v) { return v > 0.f ? v : 0.2f * v; }
__device__ __forceinline__ int   fenc(float f) { int i = __float_as_int(f); return (i >= 0) ? i : (i ^ 0x7fffffff); }
__device__ __forceinline__ float fdec(int i)   { return __int_as_float((i >= 0) ? i : (i ^ 0x7fffffff)); }

// ---------------- utility kernels ----------------
__global__ void GAT_fill_f32(float* p, float v, int n) {
  int i = blockIdx.x * blockDim.x + threadIdx.x;
  if (i < n) p[i] = v;
}
__global__ void GAT_fill_i32(int* p, int v, int n) {
  int i = blockIdx.x * blockDim.x + threadIdx.x;
  if (i < n) p[i] = v;
}
__global__ void GAT_cvt_bf16(const float* __restrict__ in, __bf16* __restrict__ out, int n) {
  int i = blockIdx.x * blockDim.x + threadIdx.x;
  if (i < n) out[i] = (__bf16)in[i];
}
// transpose-convert: in [K,N] f32 row-major -> out [N,K] bf16 (one-time weight prep)
__global__ void GAT_cvt_bf16_t(const float* __restrict__ in, __bf16* __restrict__ out, int K, int N) {
  long long idx = (long long)blockIdx.x * blockDim.x + threadIdx.x;
  if (idx >= (long long)K * N) return;
  int k = (int)(idx / N), n = (int)(idx % N);
  out[(size_t)n * K + k] = (__bf16)in[idx];
}

// ---------------- TDM 2D tile load: global -> LDS ----------------
#ifdef USE_TDM
__device__ __forceinline__ void tdm_load_2d(uint32_t lds_off, const void* gptr,
                                            uint32_t tdim0, uint32_t tdim1,
                                            uint32_t tile0, uint32_t tile1,
                                            uint64_t stride0)
{
  uint64_t ga = (uint64_t)(uintptr_t)gptr;
  u32x4 g0;
  g0[0] = 1u;                                           // count=1, user descriptor
  g0[1] = lds_off;                                      // lds_addr [63:32]
  g0[2] = (uint32_t)ga;                                 // global_addr [95:64]
  g0[3] = (uint32_t)((ga >> 32) & 0x01ffffffu)          // global_addr [120:96]
        | (2u << 30);                                   // type=2 ("image")
  i32x8 g1;
  g1[0] = (int)(1u << 16);                              // workgroup_mask=0, data_size=1 (2B)
  g1[1] = (int)((tdim0 & 0xffffu) << 16);               // tensor_dim0[15:0] @ bits63:48
  g1[2] = (int)(((tdim0 >> 16) & 0xffffu) | ((tdim1 & 0xffffu) << 16));
  g1[3] = (int)(((tdim1 >> 16) & 0xffffu) | ((tile0 & 0xffffu) << 16));   // tile_dim0
  g1[4] = (int)(tile1 & 0xffffu);                       // tile_dim1 (tile_dim2=0)
  g1[5] = (int)(uint32_t)(stride0 & 0xffffffffu);       // tensor_dim0_stride[31:0]
  g1[6] = (int)(uint32_t)((stride0 >> 32) & 0xffffu);   // stride[47:32], dim1_stride lo=0
  g1[7] = 0;
  i32x4 z4 = {0, 0, 0, 0};
  i32x8 z8 = {0, 0, 0, 0, 0, 0, 0, 0};
  // 6-arg form (clang-23 / therock-10.0 headers): (g0, g1, g2, g3, g4, cpol)
  __builtin_amdgcn_tensor_load_to_lds(g0, g1, z4, z4, z8, 0);
}
#endif

// ---------------- WMMA bf16 GEMM: C[M,N] = A[M,K] @ Bt[N,K]^T ----------------
// Block tile 64(M) x 256(N), K step 32.  8 waves: 2(M) x 4(N); each wave owns a
// 32x64 region = 2x4 v_wmma_f32_16x16x32_bf16 accumulators (8 WMMA / K-step).
#define BLK_M 64
#define BLK_N 256
#define BLK_K 32

__global__ __launch_bounds__(256) void GAT_gemm_bf16(
    const __bf16* __restrict__ A,      // [M,K]
    const __bf16* __restrict__ Bt,     // [N,K]  (weights pre-transposed)
    float* __restrict__ C, int M, int N, int K)
{
  __shared__ __bf16 As[BLK_M][BLK_K];   // 4 KB
  __shared__ __bf16 Bs[BLK_N][BLK_K];   // 16 KB, K-major rows

  const int tid  = threadIdx.x;
  const int lane = tid & 31;
  const int wave = tid >> 5;
  const int wm   = wave >> 2;               // 0..1
  const int wn   = wave & 3;                // 0..3
  const int bm   = blockIdx.x * BLK_M;
  const int bn   = blockIdx.y * BLK_N;

  v8f c[2][4] = {};

#ifdef USE_TDM
  const uint32_t ldsA = (uint32_t)(uintptr_t)&As[0][0];
  const uint32_t ldsB = (uint32_t)(uintptr_t)&Bs[0][0];
#endif

  for (int kt = 0; kt < K; kt += BLK_K) {
#ifdef USE_TDM
    if (wave == 0) {
      // A tile: 64 rows x 32 k, row stride K;  B tile: 256 rows x 32 k, row stride K
      tdm_load_2d(ldsA, A  + (size_t)bm * K + kt, (uint32_t)K, (uint32_t)M,
                  BLK_K, BLK_M, (uint64_t)K);
      tdm_load_2d(ldsB, Bt + (size_t)bn * K + kt, (uint32_t)K, (uint32_t)N,
                  BLK_K, BLK_N, (uint64_t)K);
      __builtin_amdgcn_s_wait_tensorcnt(0);
    }
    __syncthreads();
#else
    { // A tile: each thread one b128 (8 bf16)
      int row  = tid >> 2;
      int col  = (tid & 3) * 8;
      int grow = bm + row;
      uint4 v = make_uint4(0u, 0u, 0u, 0u);
      if (grow < M) {
        const __bf16* src = A + (size_t)grow * K + kt + col;
        v = *(const uint4*)src;
        if (kt + BLK_K < K) __builtin_prefetch(src + BLK_K, 0, 2);
      }
      *(uint4*)(&As[row][col]) = v;
    }
    { // B tile: each thread one full K-major row (4 x b128)
      const __bf16* src = Bt + (size_t)(bn + tid) * K + kt;
      uint4* dst = (uint4*)&Bs[tid][0];
      if (kt + BLK_K < K) __builtin_prefetch(src + BLK_K, 0, 2);
      #pragma unroll
      for (int q = 0; q < 4; ++q) dst[q] = ((const uint4*)src)[q];
    }
    __syncthreads();
#endif

    // per-lane fragments (documented 16-bit A/B VGPR layouts)
    const int mrow = lane & 15;
    const int kb   = (lane >> 4) * 8;     // A: lo lanes K0..7/16..23, hi lanes K8..15/24..31
    const int kb2  = (lane >> 4) * 8;     // B: lo lanes K0..7/16..23, hi lanes K8..15/24..31
    v16bf af[2], bfr[4];
    #pragma unroll
    for (int mi = 0; mi < 2; ++mi) {
      const __bf16* pr = &As[wm * 32 + mi * 16 + mrow][0];
      union { uint4 u[2]; v16bf v; } ua;
      ua.u[0] = *(const uint4*)(pr + kb);
      ua.u[1] = *(const uint4*)(pr + kb + 16);
      af[mi] = ua.v;
    }
    #pragma unroll
    for (int ni = 0; ni < 4; ++ni) {
      const __bf16* pr = &Bs[wn * 64 + ni * 16 + mrow][0];
      union { uint4 u[2]; v16bf v; } ub;
      ub.u[0] = *(const uint4*)(pr + kb2);
      ub.u[1] = *(const uint4*)(pr + kb2 + 16);
      bfr[ni] = ub.v;
    }

    #pragma unroll
    for (int mi = 0; mi < 2; ++mi)
      #pragma unroll
      for (int ni = 0; ni < 4; ++ni)
        c[mi][ni] = __builtin_amdgcn_wmma_f32_16x16x32_bf16(
            false, af[mi], false, bfr[ni], (short)0, c[mi][ni], false, false);
    __syncthreads();
  }

  // store C (16x16 f32 layout: VGPR r -> M = r + (lane>>4)*8, N = lane&15)
  const int crow0 = (lane >> 4) * 8;
  const int ccol  = lane & 15;
  #pragma unroll
  for (int mi = 0; mi < 2; ++mi)
    #pragma unroll
    for (int ni = 0; ni < 4; ++ni) {
      int gcol = bn + wn * 64 + ni * 16 + ccol;
      #pragma unroll
      for (int r = 0; r < 8; ++r) {
        int grow = bm + wm * 32 + mi * 16 + crow0 + r;
        if (grow < M) C[(size_t)grow * N + gcol] = c[mi][ni][r];
      }
    }
}

// ---------------- attention logits: a_s/a_d per (node, head) ----------------
__global__ void GAT_att_prep(const float* __restrict__ h,
                             const float* __restrict__ att_s,
                             const float* __restrict__ att_d,
                             float* __restrict__ as_, float* __restrict__ ad_)
{
  int idx = blockIdx.x * blockDim.x + threadIdx.x;
  if (idx >= NNODES * HEADS) return;
  int i = idx / HEADS, hd = idx - i * HEADS;
  const float* hp = h + (size_t)i * D1 + hd * OUTD;
  const float* ws = att_s + hd * OUTD;
  const float* wd = att_d + hd * OUTD;
  float ss = 0.f, sd = 0.f;
  #pragma unroll
  for (int o = 0; o < OUTD; ++o) { float v = hp[o]; ss += v * ws[o]; sd += v * wd[o]; }
  as_[idx] = ss; ad_[idx] = sd;
}

__device__ __forceinline__ void edge_sd(const int* __restrict__ ei, int e, int& s, int& d) {
  if (e < NEDGES) { s = ei[e]; d = ei[NEDGES + e]; } else { s = d = e - NEDGES; }
}

__global__ void GAT_edge_max(const int* __restrict__ ei,
                             const float* __restrict__ as_, const float* __restrict__ ad_,
                             int* __restrict__ menc)
{
  int idx = blockIdx.x * blockDim.x + threadIdx.x;
  if (idx >= ETOT * HEADS) return;
  int e = idx / HEADS, hd = idx - e * HEADS;
  int s, d; edge_sd(ei, e, s, d);
  float v = lrelu02(as_[s * HEADS + hd] + ad_[d * HEADS + hd]);
  atomicMax(&menc[d * HEADS + hd], fenc(v));
}

__global__ void GAT_edge_den(const int* __restrict__ ei,
                             const float* __restrict__ as_, const float* __restrict__ ad_,
                             const int* __restrict__ menc, float* __restrict__ den)
{
  int idx = blockIdx.x * blockDim.x + threadIdx.x;
  if (idx >= ETOT * HEADS) return;
  int e = idx / HEADS, hd = idx - e * HEADS;
  int s, d; edge_sd(ei, e, s, d);
  float v = lrelu02(as_[s * HEADS + hd] + ad_[d * HEADS + hd]);
  float m = fdec(menc[d * HEADS + hd]);
  atomicAdd(&den[d * HEADS + hd], __expf(v - m));
}

__global__ __launch_bounds__(256) void GAT_edge_agg(
    const int* __restrict__ ei,
    const float* __restrict__ as_, const float* __restrict__ ad_,
    const int* __restrict__ menc, const float* __restrict__ den,
    const float* __restrict__ h, float* __restrict__ acc)
{
  __shared__ float sAlpha[HEADS];
  int e = blockIdx.x;
  int s, d; edge_sd(ei, e, s, d);
  int tid = threadIdx.x;
  if (tid < HEADS) {
    int hd = tid;
    float v = lrelu02(as_[s * HEADS + hd] + ad_[d * HEADS + hd]);
    float m = fdec(menc[d * HEADS + hd]);
    sAlpha[hd] = __expf(v - m) / den[d * HEADS + hd];
  }
  __syncthreads();
  const float* hs = h   + (size_t)s * D1;
  float*       ac = acc + (size_t)d * D1;
  for (int el = tid; el < D1; el += 256)
    atomicAdd(&ac[el], hs[el] * sAlpha[el >> 5]);
}

__global__ void GAT_epi1(const float* __restrict__ acc, const float* __restrict__ b1,
                         const float* __restrict__ g, const float* __restrict__ bb,
                         const float* __restrict__ mu, const float* __restrict__ var,
                         __bf16* __restrict__ out)
{
  long long idx = (long long)blockIdx.x * blockDim.x + threadIdx.x;
  if (idx >= (long long)NNODES * D1) return;
  int j = (int)(idx % D1);
  float v = acc[idx] + b1[j];
  v = v > 0.f ? v : (__expf(v) - 1.f);                       // ELU
  v = (v - mu[j]) * rsqrtf(var[j] + EPS_BN) * g[j] + bb[j];  // BN eval
  out[idx] = (__bf16)v;
}

__global__ void GAT_epi2(const float* __restrict__ acc, const float* __restrict__ b2,
                         const float* __restrict__ g, const float* __restrict__ bb,
                         const float* __restrict__ mu, const float* __restrict__ var,
                         float* __restrict__ out)
{
  int idx = blockIdx.x * blockDim.x + threadIdx.x;
  if (idx >= NNODES * OUTD) return;
  int i = idx / OUTD, o = idx - i * OUTD;
  const float* a = acc + (size_t)i * D1 + o;
  float s = 0.f;
  #pragma unroll 8
  for (int hd = 0; hd < HEADS; ++hd) s += a[hd * OUTD];
  float v = s * (1.f / HEADS) + b2[o];
  v = (v - mu[o]) * rsqrtf(var[o] + EPS_BN) * g[o] + bb[o];
  out[idx] = v;
}

__global__ void GAT_pool_feat(const float* __restrict__ hfin, const int* __restrict__ batch,
                              float* __restrict__ pooled)
{
  int idx = blockIdx.x * blockDim.x + threadIdx.x;
  if (idx >= NNODES * OUTD) return;
  int i = idx / OUTD, o = idx - i * OUTD;
  atomicAdd(&pooled[batch[i] * OUTD + o], hfin[idx]);
}
__global__ void GAT_pool_cnt(const int* __restrict__ batch, float* __restrict__ cnt)
{
  int i = blockIdx.x * blockDim.x + threadIdx.x;
  if (i < NNODES) atomicAdd(&cnt[batch[i]], 1.f);
}
__global__ void GAT_final(const float* __restrict__ pooled, const float* __restrict__ cnt,
                          const float* __restrict__ w, const float* __restrict__ b,
                          float* __restrict__ out)
{
  int idx = blockIdx.x * blockDim.x + threadIdx.x;
  if (idx >= NGRAPH * 2) return;
  int gph = idx >> 1, c = idx & 1;
  float inv = 1.f / fmaxf(cnt[gph], 1.f);
  float s = b[c];
  #pragma unroll
  for (int o = 0; o < OUTD; ++o) s += pooled[gph * OUTD + o] * inv * w[o * 2 + c];
  out[idx] = s;
}

// ---------------- launch ----------------
extern "C" void kernel_launch(void* const* d_in, const int* in_sizes, int n_in,
                              void* d_out, int out_size, void* d_ws, size_t ws_size,
                              hipStream_t stream)
{
  (void)in_sizes; (void)n_in; (void)out_size; (void)ws_size;
  const float* x      = (const float*)d_in[0];
  const int*   ei     = (const int*)  d_in[1];
  const int*   batch  = (const int*)  d_in[2];
  const float* W1     = (const float*)d_in[3];
  const float* att_s1 = (const float*)d_in[4];
  const float* att_d1 = (const float*)d_in[5];
  const float* b1     = (const float*)d_in[6];
  const float* bn1_g  = (const float*)d_in[7];
  const float* bn1_b  = (const float*)d_in[8];
  const float* bn1_m  = (const float*)d_in[9];
  const float* bn1_v  = (const float*)d_in[10];
  const float* W2     = (const float*)d_in[11];
  const float* att_s2 = (const float*)d_in[12];
  const float* att_d2 = (const float*)d_in[13];
  const float* b2     = (const float*)d_in[14];
  const float* bn2_g  = (const float*)d_in[15];
  const float* bn2_b  = (const float*)d_in[16];
  const float* bn2_m  = (const float*)d_in[17];
  const float* bn2_v  = (const float*)d_in[18];
  const float* lin_w  = (const float*)d_in[19];
  const float* lin_b  = (const float*)d_in[20];
  float* out = (float*)d_out;

  // workspace carve-up
  char* w = (char*)d_ws; size_t off = 0;
  auto take = [&](size_t bytes) -> void* {
    void* p = w + off; off = (off + bytes + 255) & ~(size_t)255; return p;
  };
  __bf16* xb   = (__bf16*)take((size_t)NNODES * NFEAT * 2);
  __bf16* W1t  = (__bf16*)take((size_t)NFEAT * D1 * 2);   // [D1][NFEAT]
  __bf16* W2t  = (__bf16*)take((size_t)D1 * D1 * 2);      // [D1][D1] transposed
  float*  h    = (float*) take((size_t)NNODES * D1 * 4);  // GEMM1 out, later GEMM2 out
  float*  acc  = (float*) take((size_t)NNODES * D1 * 4);  // aggregation (both layers)
  __bf16* hbn  = (__bf16*)take((size_t)NNODES * D1 * 2);  // GEMM2 input
  float*  a_s  = (float*) take((size_t)NNODES * HEADS * 4);
  float*  a_d  = (float*) take((size_t)NNODES * HEADS * 4);
  int*    menc = (int*)   take((size_t)NNODES * HEADS * 4);
  float*  den  = (float*) take((size_t)NNODES * HEADS * 4);
  float*  hfin = (float*) take((size_t)NNODES * OUTD * 4);
  float*  pooled = (float*)take((size_t)NGRAPH * OUTD * 4);
  float*  cnt  = (float*) take((size_t)NGRAPH * 4);

  const int T = 256;
  auto blks = [](long long n, int t) { return (unsigned)((n + t - 1) / t); };
  const int NEG_INF_ENC = (int)0x807FFFFF;   // fenc(-inf)

  // 0) convert activations; transpose-convert weights (one-time prep)
  GAT_cvt_bf16<<<blks((long long)NNODES * NFEAT, T), T, 0, stream>>>(x, xb, NNODES * NFEAT);
  GAT_cvt_bf16_t<<<blks((long long)NFEAT * D1, T), T, 0, stream>>>(W1, W1t, NFEAT, D1);
  GAT_cvt_bf16_t<<<blks((long long)D1 * D1, T), T, 0, stream>>>(W2, W2t, D1, D1);

  // ---------------- layer 1 ----------------
  dim3 g1((NNODES + BLK_M - 1) / BLK_M, D1 / BLK_N);
  GAT_gemm_bf16<<<g1, 256, 0, stream>>>(xb, W1t, h, NNODES, D1, NFEAT);

  GAT_att_prep<<<blks((long long)NNODES * HEADS, T), T, 0, stream>>>(h, att_s1, att_d1, a_s, a_d);
  GAT_fill_i32<<<blks((long long)NNODES * HEADS, T), T, 0, stream>>>(menc, NEG_INF_ENC, NNODES * HEADS);
  GAT_fill_f32<<<blks((long long)NNODES * HEADS, T), T, 0, stream>>>(den, 0.f, NNODES * HEADS);
  GAT_fill_f32<<<blks((long long)NNODES * D1, T), T, 0, stream>>>(acc, 0.f, NNODES * D1);

  GAT_edge_max<<<blks((long long)ETOT * HEADS, T), T, 0, stream>>>(ei, a_s, a_d, menc);
  GAT_edge_den<<<blks((long long)ETOT * HEADS, T), T, 0, stream>>>(ei, a_s, a_d, menc, den);
  GAT_edge_agg<<<ETOT, 256, 0, stream>>>(ei, a_s, a_d, menc, den, h, acc);

  GAT_epi1<<<blks((long long)NNODES * D1, T), T, 0, stream>>>(acc, b1, bn1_g, bn1_b, bn1_m, bn1_v, hbn);

  // ---------------- layer 2 ----------------
  dim3 g2((NNODES + BLK_M - 1) / BLK_M, D1 / BLK_N);
  GAT_gemm_bf16<<<g2, 256, 0, stream>>>(hbn, W2t, h, NNODES, D1, D1);

  GAT_att_prep<<<blks((long long)NNODES * HEADS, T), T, 0, stream>>>(h, att_s2, att_d2, a_s, a_d);
  GAT_fill_i32<<<blks((long long)NNODES * HEADS, T), T, 0, stream>>>(menc, NEG_INF_ENC, NNODES * HEADS);
  GAT_fill_f32<<<blks((long long)NNODES * HEADS, T), T, 0, stream>>>(den, 0.f, NNODES * HEADS);
  GAT_fill_f32<<<blks((long long)NNODES * D1, T), T, 0, stream>>>(acc, 0.f, NNODES * D1);

  GAT_edge_max<<<blks((long long)ETOT * HEADS, T), T, 0, stream>>>(ei, a_s, a_d, menc);
  GAT_edge_den<<<blks((long long)ETOT * HEADS, T), T, 0, stream>>>(ei, a_s, a_d, menc, den);
  GAT_edge_agg<<<ETOT, 256, 0, stream>>>(ei, a_s, a_d, menc, den, h, acc);

  GAT_epi2<<<blks((long long)NNODES * OUTD, T), T, 0, stream>>>(acc, b2, bn2_g, bn2_b, bn2_m, bn2_v, hfin);

  // ---------------- pooling + head ----------------
  GAT_fill_f32<<<blks(NGRAPH * OUTD, T), T, 0, stream>>>(pooled, 0.f, NGRAPH * OUTD);
  GAT_fill_f32<<<1, 64, 0, stream>>>(cnt, 0.f, NGRAPH);
  GAT_pool_feat<<<blks((long long)NNODES * OUTD, T), T, 0, stream>>>(hfin, batch, pooled);
  GAT_pool_cnt<<<blks(NNODES, T), T, 0, stream>>>(batch, cnt);
  GAT_final<<<1, 128, 0, stream>>>(pooled, cnt, lin_w, lin_b, out);
}
